// AttentionModule_39084202394083
// MI455X (gfx1250) — matrix operand order repaired and checked by reading, not compiled, not run on hardware
//
#include <hip/hip_runtime.h>
#include <hip/hip_bf16.h>
#include <stdint.h>

// ---------------------------------------------------------------------------
// Slot-attention module for MI455X (gfx1250, wave32, WMMA bf16 16x16x32).
//   - LN(inputs) -> bf16 once; k = x@wk (bf16); v stored transposed vT[b,d,j]
//     so every WMMA B-fragment is K-contiguous.
//   - All GEMMs via v_wmma_f32_16x16x32_bf16, f32 accumulate.
//   - Slot-axis softmax is lane-local (8 slots live in one lane's v8f acc).
//   - updates GEMM (dominant HBM stream: all of vT each iteration) is K-split
//     8-way and double-buffered through LDS with GLOBAL_LOAD_ASYNC_TO_LDS_B128
//     (ASYNCcnt) for deep zero-VGPR prefetch.
// ---------------------------------------------------------------------------

typedef __bf16 bf16_t;
typedef __attribute__((ext_vector_type(16))) __bf16 v16bf;
typedef __attribute__((ext_vector_type(8)))  __bf16 v8bf;
typedef __attribute__((ext_vector_type(8)))  float   v8f;

#define B_  64
#define N_  4096
#define D_  256
#define S_  8
#define H_  1024
#define ITERS_ 3
#define KSPLIT 8
#define JCHUNK (N_ / KSPLIT)   // 512 tokens per block
#define JSTEP  64              // tokens per double-buffer stage

static __device__ __forceinline__ bf16_t f2bf(float x) { return (bf16_t)x; }

static __device__ __forceinline__ float wave_sum(float v) {
#pragma unroll
  for (int o = 16; o; o >>= 1) v += __shfl_xor(v, o, 32);
  return v;
}

static __device__ __forceinline__ v8f wmma_bf16(v16bf a, v16bf b, v8f c) {
  return __builtin_amdgcn_wmma_f32_16x16x32_bf16(false, a, false, b, (short)0, c,
                                                 false, false);
}

// A fragment (16x32 bf16): lane<16 -> row l16, K [k0..k0+15]
//                          lane>=16 -> row l16, K [k0+16..k0+31]
static __device__ __forceinline__ v16bf load_a_frag(const bf16_t* __restrict__ rowp,
                                                    int k0, int hi) {
  return *(const v16bf*)(rowp + k0 + hi * 16);
}

// B fragment (32x16 bf16) from a K-major source (row = output column n):
// lane<16 -> col l16, K [k0..k0+15]; lane>=16 -> col l16, K [k0+16..k0+31]
static __device__ __forceinline__ v16bf load_b_frag(const bf16_t* __restrict__ rowp,
                                                    int k0, int hi) {
  return *(const v16bf*)(rowp + k0 + hi * 16);
}

// Issue one async 16B global->LDS copy (ASYNCcnt path, no VGPR data).
static __device__ __forceinline__ void async_b128(const bf16_t* gsrc, bf16_t* ldst) {
  uint64_t g = (uint64_t)(uintptr_t)gsrc;
  uint32_t l = (uint32_t)(uintptr_t)ldst;  // low 32 bits of LDS aperture addr
  asm volatile("global_load_async_to_lds_b128 %0, %1, off"
               :: "v"(l), "v"(g) : "memory");
}

// ---------------------------------------------------------------------------
// Weight prep: f32 [K,N] row-major -> bf16 transposed [N,K]
// ---------------------------------------------------------------------------
__global__ void wt_transpose_bf16(const float* __restrict__ w,
                                  bf16_t* __restrict__ out, int K, int Nc) {
  int idx = blockIdx.x * 256 + threadIdx.x;
  if (idx >= K * Nc) return;
  int k = idx / Nc, n = idx % Nc;
  out[(size_t)n * K + k] = f2bf(w[idx]);
}

// ---------------------------------------------------------------------------
// LayerNorm over rows of length 256 -> bf16 (one wave32 per row, 8 rows/block)
// ---------------------------------------------------------------------------
__global__ void ln_rows_bf16(const float* __restrict__ in, const float* __restrict__ g,
                             const float* __restrict__ be, bf16_t* __restrict__ out,
                             int nrows) {
  int wave = threadIdx.x >> 5, lane = threadIdx.x & 31;
  int row = blockIdx.x * 8 + wave;
  if (row >= nrows) return;
  const float* r = in + (size_t)row * D_;
  float v[8], s = 0.f, s2 = 0.f;
#pragma unroll
  for (int t = 0; t < 8; t++) {
    float x = r[t * 32 + lane];
    v[t] = x; s += x; s2 += x * x;
  }
  s = wave_sum(s); s2 = wave_sum(s2);
  float mean = s * (1.0f / D_);
  float var = s2 * (1.0f / D_) - mean * mean;
  float rstd = rsqrtf(var + 1e-5f);
  bf16_t* o = out + (size_t)row * D_;
#pragma unroll
  for (int t = 0; t < 8; t++) {
    int c = t * 32 + lane;
    o[c] = f2bf((v[t] - mean) * rstd * g[c] + be[c]);
  }
}

// LN of slots [B,8,256] -> padded bf16 [B,16,256] (rows 8..15 zero), plus
// optional raw bf16 copy (padded) for the GRU hidden-path GEMM.
__global__ void ln_slots_bf16(const float* __restrict__ slots,
                              const float* __restrict__ g, const float* __restrict__ be,
                              bf16_t* __restrict__ ln_out, bf16_t* __restrict__ raw_out) {
  int wave = threadIdx.x >> 5, lane = threadIdx.x & 31;
  int row = blockIdx.x * 8 + wave;  // [0, B*16)
  if (row >= B_ * 16) return;
  int b = row >> 4, i = row & 15;
  bf16_t* lo = ln_out + (size_t)row * D_;
  bf16_t* ro = raw_out ? raw_out + (size_t)row * D_ : nullptr;
  if (i >= S_) {
#pragma unroll
    for (int t = 0; t < 8; t++) {
      int c = t * 32 + lane;
      lo[c] = f2bf(0.f);
      if (ro) ro[c] = f2bf(0.f);
    }
    return;
  }
  const float* r = slots + ((size_t)b * S_ + i) * D_;
  float v[8], s = 0.f, s2 = 0.f;
#pragma unroll
  for (int t = 0; t < 8; t++) {
    float x = r[t * 32 + lane];
    v[t] = x; s += x; s2 += x * x;
  }
  s = wave_sum(s); s2 = wave_sum(s2);
  float mean = s * (1.0f / D_);
  float var = s2 * (1.0f / D_) - mean * mean;
  float rstd = rsqrtf(var + 1e-5f);
#pragma unroll
  for (int t = 0; t < 8; t++) {
    int c = t * 32 + lane;
    lo[c] = f2bf((v[t] - mean) * rstd * g[c] + be[c]);
    if (ro) ro[c] = f2bf(v[t]);
  }
}

// ---------------------------------------------------------------------------
// Generic GEMM: C[M,N] = A[M,K](bf16) @ W (given as Wt[N,K] bf16) + bias(f32)
// Block 256 thr = 8 waves; block tile 64x64; wave tile 16x32 (2 WMMA C tiles).
// VTRANS: scatter v into vT[b, d, j] (b = m>>12, j = m&4095), bf16.
// ---------------------------------------------------------------------------
template <bool OUT_BF16, bool RELU, bool VTRANS>
__global__ void gemm_bf16(const bf16_t* __restrict__ A, const bf16_t* __restrict__ Wt,
                          const float* __restrict__ bias, void* __restrict__ Cout,
                          int M, int Nc, int K, int ldc) {
  int lane = threadIdx.x & 31, wave = threadIdx.x >> 5;
  int l16 = lane & 15, hi = lane >> 4;
  int rowBase = blockIdx.x * 64 + (wave >> 1) * 16;
  int colBase = blockIdx.y * 64 + (wave & 1) * 32;
  const bf16_t* arow = A + (size_t)(rowBase + l16) * K;
  const bf16_t* brow0 = Wt + (size_t)(colBase + l16) * K;
  const bf16_t* brow1 = Wt + (size_t)(colBase + 16 + l16) * K;
  v8f c0 = {}, c1 = {};
  for (int k0 = 0; k0 < K; k0 += 32) {
    v16bf a = load_a_frag(arow, k0, hi);
    v16bf b0 = load_b_frag(brow0, k0, hi);
    v16bf b1 = load_b_frag(brow1, k0, hi);
    c0 = wmma_bf16(a, b0, c0);
    c1 = wmma_bf16(a, b1, c1);
  }
  int m0 = rowBase + hi * 8;
#pragma unroll
  for (int r = 0; r < 8; r++) {
    int m = m0 + r;
#pragma unroll
    for (int t = 0; t < 2; t++) {
      int n = colBase + t * 16 + l16;
      float val = (t ? c1[r] : c0[r]) + bias[n];
      if (RELU) val = fmaxf(val, 0.f);
      if (OUT_BF16) {
        if (VTRANS) {
          size_t bb = (size_t)(m >> 12);
          size_t j = (size_t)(m & 4095);
          ((bf16_t*)Cout)[(bb * D_ + n) * N_ + j] = f2bf(val);
        } else {
          ((bf16_t*)Cout)[(size_t)m * ldc + n] = f2bf(val);
        }
      } else {
        ((float*)Cout)[(size_t)m * ldc + n] = val;
      }
    }
  }
}

// ---------------------------------------------------------------------------
// dots[b,i,j] = scale * <q[b,i,:], k[b,j,:]>, softmax over slot axis (lane
// local: lanes 0-15 hold all 8 slot rows), +EPS, write p (f32).
// ---------------------------------------------------------------------------
__global__ void dots_softmax(const bf16_t* __restrict__ qbf,
                             const bf16_t* __restrict__ kbf, float* __restrict__ p) {
  int lane = threadIdx.x & 31, wave = threadIdx.x >> 5;
  int l16 = lane & 15, hi = lane >> 4;
  int b = blockIdx.y;
  int jbase = blockIdx.x * 128 + wave * 16;
  const bf16_t* arow = qbf + ((size_t)b * 16 + l16) * D_;
  const bf16_t* brow = kbf + ((size_t)b * N_ + (jbase + l16)) * D_;
  v8f c = {};
#pragma unroll
  for (int k0 = 0; k0 < D_; k0 += 32) {
    v16bf a = load_a_frag(arow, k0, hi);
    v16bf bb = load_b_frag(brow, k0, hi);
    c = wmma_bf16(a, bb, c);
  }
  if (hi == 0) {
    float d[8], mx = -3.0e38f;
#pragma unroll
    for (int r = 0; r < 8; r++) { d[r] = c[r] * 0.0625f; mx = fmaxf(mx, d[r]); }
    float sum = 0.f;
#pragma unroll
    for (int r = 0; r < 8; r++) { d[r] = expf(d[r] - mx); sum += d[r]; }
    float inv = 1.0f / sum;
    int j = jbase + l16;
#pragma unroll
    for (int r = 0; r < 8; r++)
      p[((size_t)b * S_ + r) * N_ + j] = d[r] * inv + 1e-8f;
  }
}

// Normalize attn over token axis, emit padded bf16 [B,16,N] (rows 8..15 zero)
__global__ void attn_norm(const float* __restrict__ p, bf16_t* __restrict__ attn) {
  int b = blockIdx.x >> 4, i = blockIdx.x & 15;
  bf16_t* orow = attn + ((size_t)b * 16 + i) * N_;
  if (i >= S_) {
#pragma unroll
    for (int t = 0; t < 16; t++) orow[t * 256 + threadIdx.x] = f2bf(0.f);
    return;
  }
  const float* pr = p + ((size_t)b * S_ + i) * N_;
  float v[16], s = 0.f;
#pragma unroll
  for (int t = 0; t < 16; t++) { v[t] = pr[t * 256 + threadIdx.x]; s += v[t]; }
  s = wave_sum(s);
  __shared__ float ws8[8];
  if ((threadIdx.x & 31) == 0) ws8[threadIdx.x >> 5] = s;
  __syncthreads();
  float tot = 0.f;
#pragma unroll
  for (int w = 0; w < 8; w++) tot += ws8[w];
  float inv = 1.0f / tot;
#pragma unroll
  for (int t = 0; t < 16; t++) orow[t * 256 + threadIdx.x] = f2bf(v[t] * inv);
}

// ---------------------------------------------------------------------------
// updates partial GEMM, K-split 8-way, async LDS double buffering.
// Block (b, kc): M=16 slots x N=256 dims over j in [kc*512, kc*512+512).
// Stage = 64 tokens: B tile 256x64 bf16 (32KB) + A tile 16x64 bf16 (2KB),
// fetched with GLOBAL_LOAD_ASYNC_TO_LDS_B128; per wave exactly 9 async ops
// per stage (8 full-EXEC for B, 1 half-EXEC for A) so `s_wait_asynccnt 9`
// after issuing stage s+1 precisely drains stage s (in-order completion).
// ---------------------------------------------------------------------------
__global__ void updates_gemm_async(const bf16_t* __restrict__ attn,
                                   const bf16_t* __restrict__ vT,
                                   float* __restrict__ updp) {
  __shared__ bf16_t sB[2][D_ * JSTEP];   // 2 x 32KB
  __shared__ bf16_t sA[2][16 * JSTEP];   // 2 x 2KB
  int tid = threadIdx.x;
  int lane = tid & 31, wave = tid >> 5;
  int l16 = lane & 15, hi = lane >> 4;
  int b = blockIdx.x;
  int kc = blockIdx.y;
  int j0 = kc * JCHUNK;
  const bf16_t* vbase = vT + (size_t)b * D_ * N_;
  const bf16_t* abase = attn + (size_t)b * 16 * N_;

  auto issue = [&](int s, int buf) {
    int jb = j0 + s * JSTEP;
    // B tile: 2048 x 16B chunks, 8 per thread (full EXEC -> 8 ops/wave)
#pragma unroll
    for (int c = 0; c < 8; ++c) {
      int u = c * 256 + tid;   // 0..2047
      int d = u >> 3, q = u & 7;
      async_b128(vbase + (size_t)d * N_ + jb + q * 8, &sB[buf][d * JSTEP + q * 8]);
    }
    // A tile: 128 x 16B chunks spread as wave*16 + l16, lanes 16..31 idle
    // (one instruction per wave regardless -> uniform ASYNCcnt).
    {
      int u = wave * 16 + l16;  // 0..127
      int r = u >> 3, q = u & 7;
      if (lane < 16)
        async_b128(abase + (size_t)r * N_ + jb + q * 8, &sA[buf][r * JSTEP + q * 8]);
    }
  };

  issue(0, 0);
  v8f c0 = {}, c1 = {};
  int colBase = wave * 32;
  const int NSTAGE = JCHUNK / JSTEP;  // 8
  for (int s = 0; s < NSTAGE; ++s) {
    int buf = s & 1;
    if (s + 1 < NSTAGE) {
      issue(s + 1, buf ^ 1);
      asm volatile("s_wait_asynccnt 9" ::: "memory");   // stage s complete
    } else {
      asm volatile("s_wait_asynccnt 0" ::: "memory");
    }
    __syncthreads();
    const bf16_t* At = &sA[buf][(size_t)l16 * JSTEP];
    const bf16_t* Bt0 = &sB[buf][(size_t)(colBase + l16) * JSTEP];
    const bf16_t* Bt1 = &sB[buf][(size_t)(colBase + 16 + l16) * JSTEP];
#pragma unroll
    for (int kk = 0; kk < JSTEP; kk += 32) {
      v16bf a = load_a_frag(At, kk, hi);
      v16bf b0 = load_b_frag(Bt0, kk, hi);
      v16bf b1 = load_b_frag(Bt1, kk, hi);
      c0 = wmma_bf16(a, b0, c0);
      c1 = wmma_bf16(a, b1, c1);
    }
    __syncthreads();
  }
  float* out = updp + ((size_t)kc * B_ + b) * 16 * D_;
  int m0 = hi * 8;
#pragma unroll
  for (int r = 0; r < 8; r++) {
#pragma unroll
    for (int t = 0; t < 2; t++) {
      int n = colBase + t * 16 + l16;
      out[(size_t)(m0 + r) * D_ + n] = (t ? c1[r] : c0[r]);
    }
  }
}

// Reduce K-split partials -> updates bf16 [B,16,256]
__global__ void updates_reduce(const float* __restrict__ updp,
                               bf16_t* __restrict__ updbf) {
  int idx = blockIdx.x * 256 + threadIdx.x;  // over B*16*256
  float s = 0.f;
#pragma unroll
  for (int kc = 0; kc < KSPLIT; ++kc)
    s += updp[(size_t)kc * (B_ * 16 * D_) + idx];
  updbf[idx] = f2bf(s);
}

// GRU cell combine (element-wise over B*8*256)
__global__ void gru_update(const float* __restrict__ gx, const float* __restrict__ gh,
                           float* __restrict__ slots) {
  int idx = blockIdx.x * 256 + threadIdx.x;
  int d = idx & 255, i = (idx >> 8) & 7, b = idx >> 11;
  const float* gxr = gx + ((size_t)b * 16 + i) * 768;
  const float* ghr = gh + ((size_t)b * 16 + i) * 768;
  float r = 1.f / (1.f + expf(-(gxr[d] + ghr[d])));
  float z = 1.f / (1.f + expf(-(gxr[256 + d] + ghr[256 + d])));
  float nn = tanhf(gxr[512 + d] + r * ghr[512 + d]);
  float prev = slots[idx];
  slots[idx] = (1.f - z) * nn + z * prev;
}

__global__ void add_residual(const float* __restrict__ mo, float* __restrict__ slots) {
  int idx = blockIdx.x * 256 + threadIdx.x;
  int d = idx & 255, i = (idx >> 8) & 7, b = idx >> 11;
  slots[idx] += mo[((size_t)b * 16 + i) * D_ + d];
}

// slots [S,B,D] -> internal [B,S,D]
__global__ void slots_in_kernel(const float* __restrict__ sin, float* __restrict__ slots) {
  int idx = blockIdx.x * 256 + threadIdx.x;
  int d = idx & 255, s = (idx >> 8) & 7, b = idx >> 11;
  slots[idx] = sin[((size_t)s * B_ + b) * D_ + d];
}
// internal [B,S,D] -> out [S,B,D]
__global__ void slots_out_kernel(const float* __restrict__ slots, float* __restrict__ out) {
  int idx = blockIdx.x * 256 + threadIdx.x;
  int d = idx & 255, s = (idx >> 8) & 7, b = idx >> 11;
  out[((size_t)s * B_ + b) * D_ + d] = slots[idx];
}

// ---------------------------------------------------------------------------
extern "C" void kernel_launch(void* const* d_in, const int* in_sizes, int n_in,
                              void* d_out, int out_size, void* d_ws, size_t ws_size,
                              hipStream_t stream) {
  (void)in_sizes; (void)n_in; (void)out_size; (void)ws_size;
  const float* inputs = (const float*)d_in[0];
  const float* slots0 = (const float*)d_in[1];
  const float* wq = (const float*)d_in[2];
  const float* bq = (const float*)d_in[3];
  const float* wk = (const float*)d_in[4];
  const float* bk = (const float*)d_in[5];
  const float* wv = (const float*)d_in[6];
  const float* bv = (const float*)d_in[7];
  const float* w_ih = (const float*)d_in[8];
  const float* b_ih = (const float*)d_in[9];
  const float* w_hh = (const float*)d_in[10];
  const float* b_hh = (const float*)d_in[11];
  const float* mw1 = (const float*)d_in[12];
  const float* mb1 = (const float*)d_in[13];
  const float* mw2 = (const float*)d_in[14];
  const float* mb2 = (const float*)d_in[15];
  const float* g_in = (const float*)d_in[16];
  const float* b_in = (const float*)d_in[17];
  const float* g_sl = (const float*)d_in[18];
  const float* b_sl = (const float*)d_in[19];
  const float* g_ff = (const float*)d_in[20];
  const float* b_ff = (const float*)d_in[21];

  uint8_t* base = (uint8_t*)d_ws;
  size_t off = 0;
  auto alloc = [&](size_t bytes) -> void* {
    void* pp = base + off;
    off = (off + bytes + 255) & ~(size_t)255;
    return pp;
  };
  bf16_t* xbf   = (bf16_t*)alloc((size_t)B_ * N_ * D_ * 2);
  bf16_t* kbf   = (bf16_t*)alloc((size_t)B_ * N_ * D_ * 2);
  bf16_t* vT    = (bf16_t*)alloc((size_t)B_ * D_ * N_ * 2);
  bf16_t* wq_t  = (bf16_t*)alloc((size_t)D_ * D_ * 2);
  bf16_t* wk_t  = (bf16_t*)alloc((size_t)D_ * D_ * 2);
  bf16_t* wv_t  = (bf16_t*)alloc((size_t)D_ * D_ * 2);
  bf16_t* wih_t = (bf16_t*)alloc((size_t)D_ * 768 * 2);
  bf16_t* whh_t = (bf16_t*)alloc((size_t)D_ * 768 * 2);
  bf16_t* w1_t  = (bf16_t*)alloc((size_t)D_ * H_ * 2);
  bf16_t* w2_t  = (bf16_t*)alloc((size_t)H_ * D_ * 2);
  float*  slots = (float*)alloc((size_t)B_ * S_ * D_ * 4);
  bf16_t* sn_bf = (bf16_t*)alloc((size_t)B_ * 16 * D_ * 2);
  bf16_t* raw_bf= (bf16_t*)alloc((size_t)B_ * 16 * D_ * 2);
  bf16_t* q_bf  = (bf16_t*)alloc((size_t)B_ * 16 * D_ * 2);
  float*  pbuf  = (float*)alloc((size_t)B_ * S_ * N_ * 4);
  bf16_t* attn  = (bf16_t*)alloc((size_t)B_ * 16 * N_ * 2);
  float*  updp  = (float*)alloc((size_t)KSPLIT * B_ * 16 * D_ * 4);
  bf16_t* updbf = (bf16_t*)alloc((size_t)B_ * 16 * D_ * 2);
  float*  gx    = (float*)alloc((size_t)B_ * 16 * 768 * 4);
  float*  gh    = (float*)alloc((size_t)B_ * 16 * 768 * 4);
  bf16_t* ff_bf = (bf16_t*)alloc((size_t)B_ * 16 * D_ * 2);
  bf16_t* h1_bf = (bf16_t*)alloc((size_t)B_ * 16 * H_ * 2);
  float*  mlpo  = (float*)alloc((size_t)B_ * 16 * D_ * 4);

  const int tpb = 256;
  // Weight prep (bf16, transposed to [N,K])
  wt_transpose_bf16<<<(D_ * D_ + 255) / 256, tpb, 0, stream>>>(wq, wq_t, D_, D_);
  wt_transpose_bf16<<<(D_ * D_ + 255) / 256, tpb, 0, stream>>>(wk, wk_t, D_, D_);
  wt_transpose_bf16<<<(D_ * D_ + 255) / 256, tpb, 0, stream>>>(wv, wv_t, D_, D_);
  wt_transpose_bf16<<<(D_ * 768 + 255) / 256, tpb, 0, stream>>>(w_ih, wih_t, D_, 768);
  wt_transpose_bf16<<<(D_ * 768 + 255) / 256, tpb, 0, stream>>>(w_hh, whh_t, D_, 768);
  wt_transpose_bf16<<<(D_ * H_ + 255) / 256, tpb, 0, stream>>>(mw1, w1_t, D_, H_);
  wt_transpose_bf16<<<(H_ * D_ + 255) / 256, tpb, 0, stream>>>(mw2, w2_t, H_, D_);
  slots_in_kernel<<<(B_ * S_ * D_) / 256, tpb, 0, stream>>>(slots0, slots);

  // x = LN(inputs) -> bf16; k = x@wk (bf16); v = x@wv stored transposed
  ln_rows_bf16<<<(B_ * N_) / 8, tpb, 0, stream>>>(inputs, g_in, b_in, xbf, B_ * N_);
  gemm_bf16<true, false, false><<<dim3((B_ * N_) / 64, D_ / 64), tpb, 0, stream>>>(
      xbf, wk_t, bk, kbf, B_ * N_, D_, D_, D_);
  gemm_bf16<true, false, true><<<dim3((B_ * N_) / 64, D_ / 64), tpb, 0, stream>>>(
      xbf, wv_t, bv, vT, B_ * N_, D_, D_, D_);

  for (int it = 0; it < ITERS_; ++it) {
    ln_slots_bf16<<<(B_ * 16) / 8, tpb, 0, stream>>>(slots, g_sl, b_sl, sn_bf, raw_bf);
    gemm_bf16<true, false, false><<<dim3(16, 4), tpb, 0, stream>>>(
        sn_bf, wq_t, bq, q_bf, B_ * 16, D_, D_, D_);
    dots_softmax<<<dim3(N_ / 128, B_), tpb, 0, stream>>>(q_bf, kbf, pbuf);
    attn_norm<<<B_ * 16, tpb, 0, stream>>>(pbuf, attn);
    updates_gemm_async<<<dim3(B_, KSPLIT), tpb, 0, stream>>>(attn, vT, updp);
    updates_reduce<<<(B_ * 16 * D_) / 256, tpb, 0, stream>>>(updp, updbf);
    gemm_bf16<false, false, false><<<dim3(16, 12), tpb, 0, stream>>>(
        updbf, wih_t, b_ih, gx, B_ * 16, 768, D_, 768);
    gemm_bf16<false, false, false><<<dim3(16, 12), tpb, 0, stream>>>(
        raw_bf, whh_t, b_hh, gh, B_ * 16, 768, D_, 768);
    gru_update<<<(B_ * S_ * D_) / 256, tpb, 0, stream>>>(gx, gh, slots);
    ln_slots_bf16<<<(B_ * 16) / 8, tpb, 0, stream>>>(slots, g_ff, b_ff, ff_bf, nullptr);
    gemm_bf16<true, true, false><<<dim3(16, 16), tpb, 0, stream>>>(
        ff_bf, w1_t, mb1, h1_bf, B_ * 16, H_, D_, H_);
    gemm_bf16<false, false, false><<<dim3(16, 4), tpb, 0, stream>>>(
        h1_bf, w2_t, mb2, mlpo, B_ * 16, D_, H_, D_);
    add_residual<<<(B_ * S_ * D_) / 256, tpb, 0, stream>>>(mlpo, slots);
  }
  slots_out_kernel<<<(B_ * S_ * D_) / 256, tpb, 0, stream>>>(slots, (float*)d_out);
}
